// MultiHeadLatentAttention_30906584662371
// MI455X (gfx1250) — compile-verified, matrix operand order checked by reading, not run on hardware
//
#include <hip/hip_runtime.h>
#include <hip/hip_bf16.h>
#include <math.h>

#define BDIM   4
#define SEQ    2048
#define NHEADS 16
#define DMODEL 1024
#define DLAT   256
#define DH     64
#define DHR    32
#define DQK    96
#define MROWS  (BDIM*SEQ)   // 8192

typedef __attribute__((ext_vector_type(16))) __bf16          v16bf;
typedef __attribute__((ext_vector_type(8)))  float           v8f;
typedef __attribute__((ext_vector_type(16))) unsigned short  v16u;
typedef __attribute__((ext_vector_type(8)))  unsigned short  v8u;

__device__ __forceinline__ unsigned short f2bf(float f) {
  union { float f; unsigned int u; } c; c.f = f;
  unsigned int u = c.u;
  unsigned int r = u + 0x7FFFu + ((u >> 16) & 1u);   // round to nearest even
  return (unsigned short)(r >> 16);
}

// Build a 16x32 (bf16) WMMA A/B operand fragment from two contiguous 8-element
// bf16 runs. Per CDNA5 ISA packing: elem e in [0,8): k = e + 8*half;
// elem e in [8,16): k = (e-8) + 16 + 8*half.
__device__ __forceinline__ v16bf frag_from(const unsigned short* lo,
                                           const unsigned short* hi) {
  v8u a = *(const v8u*)lo;
  v8u b = *(const v8u*)hi;
  v16u t;
#pragma unroll
  for (int e = 0; e < 8; ++e) { t[e] = a[e]; t[8 + e] = b[e]; }
  return __builtin_bit_cast(v16bf, t);
}

// ---------------------------------------------------------------------------
// One-time fp32 -> bf16 conversions
// ---------------------------------------------------------------------------
__global__ void cast_bf16_kernel(const float* __restrict__ src,
                                 unsigned short* __restrict__ dst, size_t n)
{
  size_t i = (size_t)blockIdx.x * blockDim.x + threadIdx.x;
  if (i < n) dst[i] = f2bf(src[i]);
}

// dst[n][k] = bf16(src[k][n])   (weights: [K][N] -> transposed bf16 [N][K])
__global__ void castT_bf16_kernel(const float* __restrict__ src,
                                  unsigned short* __restrict__ dst, int K, int N)
{
  size_t i = (size_t)blockIdx.x * blockDim.x + threadIdx.x;
  size_t total = (size_t)K * N;
  if (i >= total) return;
  int k = (int)(i % K);
  int n = (int)(i / K);
  dst[i] = f2bf(src[(size_t)k * N + n]);
}

// ---------------------------------------------------------------------------
// GEMM: C[M,N] = A[M,K] * B[K,N]; A bf16 row-major, BT = bf16 B^T [N][K].
// 128 threads = 4 waves; block tile 128x64; wave tile 32x64 (2x4 WMMA tiles).
// No LDS: fragments loaded straight from global (192MB L2 holds everything).
// Requires M % 128 == 0, K % 32 == 0; N guarded. outBF16: 1 -> bf16 C.
// ---------------------------------------------------------------------------
#define GEMM_BM 128
#define GEMM_BN 64

__global__ __launch_bounds__(128)
void gemm_bf16_kernel(const unsigned short* __restrict__ A,
                      const unsigned short* __restrict__ BT,
                      void* __restrict__ Cout, int M, int N, int K, int outBF16)
{
  const int wave = threadIdx.x >> 5;
  const int lane = threadIdx.x & 31;
  const int r = lane & 15;
  const int h = lane >> 4;
  const int mBase = blockIdx.y * GEMM_BM + wave * 32;
  const int nBase = blockIdx.x * GEMM_BN;

  const unsigned short* a0 = A + (size_t)(mBase + r) * K;
  const unsigned short* a1 = A + (size_t)(mBase + 16 + r) * K;
  const unsigned short* brow[4];
#pragma unroll
  for (int t = 0; t < 4; ++t) {
    int gn = nBase + t * 16 + r;
    int cl = gn < N ? gn : (N - 1);      // clamp; garbage discarded at store
    brow[t] = BT + (size_t)cl * K;
  }

  v8f acc[2][4];
#pragma unroll
  for (int mi = 0; mi < 2; ++mi)
#pragma unroll
    for (int t = 0; t < 4; ++t) acc[mi][t] = v8f{};

  for (int k0 = 0; k0 < K; k0 += 32) {
    __builtin_prefetch(a0 + k0 + 256, 0, 1);   // -> global_prefetch_b8
    __builtin_prefetch(brow[0] + k0 + 256, 0, 1);
    v16bf af0 = frag_from(a0 + k0 + 8 * h, a0 + k0 + 16 + 8 * h);
    v16bf af1 = frag_from(a1 + k0 + 8 * h, a1 + k0 + 16 + 8 * h);
    v16bf bfr[4];
#pragma unroll
    for (int t = 0; t < 4; ++t)
      bfr[t] = frag_from(brow[t] + k0 + 8 * h, brow[t] + k0 + 16 + 8 * h);
#pragma unroll
    for (int t = 0; t < 4; ++t) {
      acc[0][t] = __builtin_amdgcn_wmma_f32_16x16x32_bf16(
          false, af0, false, bfr[t], (short)0, acc[0][t], false, false);
      acc[1][t] = __builtin_amdgcn_wmma_f32_16x16x32_bf16(
          false, af1, false, bfr[t], (short)0, acc[1][t], false, false);
    }
  }

  // C/D layout: lane n = lane%16, VGPR v -> m = v + 8*(lane/16)
#pragma unroll
  for (int mi = 0; mi < 2; ++mi) {
    const int m0 = mBase + mi * 16 + 8 * h;
#pragma unroll
    for (int t = 0; t < 4; ++t) {
      int gn = nBase + t * 16 + r;
      if (gn < N) {
        if (outBF16) {
          unsigned short* C = (unsigned short*)Cout;
#pragma unroll
          for (int v = 0; v < 8; ++v)
            C[(size_t)(m0 + v) * N + gn] = f2bf(acc[mi][t][v]);
        } else {
          float* C = (float*)Cout;
#pragma unroll
          for (int v = 0; v < 8; ++v)
            C[(size_t)(m0 + v) * N + gn] = acc[mi][t][v];
        }
      }
    }
  }
}

// ---------------------------------------------------------------------------
// Pack kernels: rope + concat into bf16 attention operands
// ---------------------------------------------------------------------------
__global__ void pack_q_kernel(const float* __restrict__ Qc, const float* __restrict__ Qr,
                              unsigned short* __restrict__ Qb)
{
  size_t idx = (size_t)blockIdx.x * blockDim.x + threadIdx.x;
  const size_t total = (size_t)BDIM * NHEADS * SEQ * DQK;
  if (idx >= total) return;
  int d  = (int)(idx % DQK);
  int s  = (int)((idx / DQK) % SEQ);
  int hh = (int)((idx / ((size_t)DQK * SEQ)) % NHEADS);
  int b  = (int)(idx / ((size_t)DQK * SEQ * NHEADS));
  size_t row = (size_t)b * SEQ + s;
  float val;
  if (d < DH) {
    val = Qc[row * (NHEADS * DH) + hh * DH + d];
  } else {
    int dr = d - DH;
    int i = dr >> 1, comp = dr & 1;
    float inv = __powf(10000.0f, -(float)(2 * i) / (float)DHR);
    float sn, cs; __sincosf((float)s * inv, &sn, &cs);
    float x1 = Qr[row * (NHEADS * DHR) + hh * DHR + 2 * i];
    float x2 = Qr[row * (NHEADS * DHR) + hh * DHR + 2 * i + 1];
    val = comp ? (x1 * sn + x2 * cs) : (x1 * cs - x2 * sn);
  }
  Qb[idx] = f2bf(val);
}

__global__ void pack_k_kernel(const float* __restrict__ Kc, const float* __restrict__ Krp,
                              unsigned short* __restrict__ Kb)
{
  size_t idx = (size_t)blockIdx.x * blockDim.x + threadIdx.x;
  const size_t total = (size_t)BDIM * NHEADS * SEQ * DQK;
  if (idx >= total) return;
  int d  = (int)(idx % DQK);
  int s  = (int)((idx / DQK) % SEQ);
  int hh = (int)((idx / ((size_t)DQK * SEQ)) % NHEADS);
  int b  = (int)(idx / ((size_t)DQK * SEQ * NHEADS));
  size_t row = (size_t)b * SEQ + s;
  float val;
  if (d < DH) {
    val = Kc[row * (NHEADS * DH) + hh * DH + d];
  } else {
    int dr = d - DH;                 // Kr broadcast over heads
    int i = dr >> 1, comp = dr & 1;
    float inv = __powf(10000.0f, -(float)(2 * i) / (float)DHR);
    float sn, cs; __sincosf((float)s * inv, &sn, &cs);
    float x1 = Krp[row * DHR + 2 * i];
    float x2 = Krp[row * DHR + 2 * i + 1];
    val = comp ? (x1 * sn + x2 * cs) : (x1 * cs - x2 * sn);
  }
  Kb[idx] = f2bf(val);
}

__global__ void pack_vt_kernel(const float* __restrict__ V, unsigned short* __restrict__ Vt)
{
  size_t idx = (size_t)blockIdx.x * blockDim.x + threadIdx.x;
  const size_t total = (size_t)BDIM * NHEADS * DH * SEQ;
  if (idx >= total) return;
  int s  = (int)(idx % SEQ);
  int d  = (int)((idx / SEQ) % DH);
  int hh = (int)((idx / ((size_t)SEQ * DH)) % NHEADS);
  int b  = (int)(idx / ((size_t)SEQ * DH * NHEADS));
  Vt[idx] = f2bf(V[((size_t)b * SEQ + s) * (NHEADS * DH) + hh * DH + d]);
}

// ---------------------------------------------------------------------------
// Flash attention. One wave per 16-row q tile; S^T = K_tile x Q^T so softmax
// reductions are per-lane, and S^T C-tiles map register-for-register into the
// B operand of O^T = V^T x P^T. Output written as bf16 row-major [B*S][1024]
// to feed the W_O GEMM directly.
// ---------------------------------------------------------------------------
__global__ __launch_bounds__(128)
void mla_attention_kernel(const unsigned short* __restrict__ Qb,
                          const unsigned short* __restrict__ Kb,
                          const unsigned short* __restrict__ Vt,
                          unsigned short* __restrict__ attn)
{
  const int wave = threadIdx.x >> 5;
  const int lane = threadIdx.x & 31;
  const int r = lane & 15;
  const int h = lane >> 4;
  const int QTILES = SEQ / 16;                 // 128
  int task = blockIdx.x * 4 + wave;            // B*H*QTILES = 8192 tasks
  int qt = task % QTILES;
  int bh = task / QTILES;                      // b*NHEADS + head

  const unsigned short* Qp = Qb + (size_t)bh * SEQ * DQK;
  const unsigned short* Kp = Kb + (size_t)bh * SEQ * DQK;
  const unsigned short* Vp = Vt + (size_t)bh * DH * SEQ;

  const int q = qt * 16 + r;

  // Q fragments (B operand of S^T), fixed for the whole k loop: 3 chunks of d
  v16bf qf[3];
#pragma unroll
  for (int c = 0; c < 3; ++c) {
    const unsigned short* row = Qp + (size_t)q * DQK + 32 * c;
    qf[c] = frag_from(row + 8 * h, row + 16 + 8 * h);
  }

  v8f Ot[4] = {v8f{}, v8f{}, v8f{}, v8f{}};    // O^T: lane=q, tile dt rows d=16dt+v+8h
  float Mv = -INFINITY, Lv = 0.0f;
  const float scale = 0.10206207261596575f;    // 1/sqrt(96)

  const int kbMax = (qt * 16 + 15) >> 5;       // inclusive, causal
  for (int kb = 0; kb <= kbMax; ++kb) {
    v8f sc0 = {}, sc1 = {};
#pragma unroll
    for (int c = 0; c < 3; ++c) {
      const unsigned short* k0 = Kp + (size_t)(kb * 32 + r) * DQK + 32 * c;
      v16bf kf0 = frag_from(k0 + 8 * h, k0 + 16 + 8 * h);
      sc0 = __builtin_amdgcn_wmma_f32_16x16x32_bf16(false, kf0, false, qf[c],
                                                    (short)0, sc0, false, false);
      const unsigned short* k1 = Kp + (size_t)(kb * 32 + 16 + r) * DQK + 32 * c;
      v16bf kf1 = frag_from(k1 + 8 * h, k1 + 16 + 8 * h);
      sc1 = __builtin_amdgcn_wmma_f32_16x16x32_bf16(false, kf1, false, qf[c],
                                                    (short)0, sc1, false, false);
    }

    // clip, scale, causal mask; local max over the 16 k-values this lane holds
    float vals[16];
    float mloc = -INFINITY;
#pragma unroll
    for (int v = 0; v < 8; ++v) {
      int k0i = kb * 32 + v + 8 * h;
      float s0 = fminf(fmaxf(sc0[v], -80.0f), 80.0f) * scale;
      s0 = (k0i <= q) ? s0 : -INFINITY;
      vals[v] = s0; mloc = fmaxf(mloc, s0);
      int k1i = kb * 32 + 16 + v + 8 * h;
      float s1 = fminf(fmaxf(sc1[v], -80.0f), 80.0f) * scale;
      s1 = (k1i <= q) ? s1 : -INFINITY;
      vals[8 + v] = s1; mloc = fmaxf(mloc, s1);
    }
    mloc = fmaxf(mloc, __shfl_xor(mloc, 16, 32));   // combine lane halves of row q
    float Mnew = fmaxf(Mv, mloc);
    float f = __expf(Mv - Mnew);                    // 0 on first block
    float sloc = 0.0f;
    v16u pf;
#pragma unroll
    for (int i = 0; i < 16; ++i) {
      float p = __expf(vals[i] - Mnew);
      sloc += p;
      pf[i] = f2bf(p);
    }
    sloc += __shfl_xor(sloc, 16, 32);
    Lv = Lv * f + sloc;
    Mv = Mnew;
#pragma unroll
    for (int dt = 0; dt < 4; ++dt)
#pragma unroll
      for (int v = 0; v < 8; ++v) Ot[dt][v] *= f;   // f is per-lane(q) uniform

    // P^T B-operand == direct register remap of the two S^T C-tiles
    v16bf pbf = __builtin_bit_cast(v16bf, pf);
#pragma unroll
    for (int dt = 0; dt < 4; ++dt) {
      const unsigned short* vrow = Vp + (size_t)(16 * dt + r) * SEQ + kb * 32;
      v16bf vf = frag_from(vrow + 8 * h, vrow + 16 + 8 * h);
      Ot[dt] = __builtin_amdgcn_wmma_f32_16x16x32_bf16(false, vf, false, pbf,
                                                       (short)0, Ot[dt], false, false);
    }
  }

  float invL = 1.0f / Lv;
  int b = bh >> 4, head = bh & 15;
  unsigned short* op = attn + ((size_t)b * SEQ + q) * (NHEADS * DH) + head * DH;
#pragma unroll
  for (int dt = 0; dt < 4; ++dt)
#pragma unroll
    for (int v = 0; v < 8; ++v)
      op[16 * dt + v + 8 * h] = f2bf(Ot[dt][v] * invL);
}

// ---------------------------------------------------------------------------
extern "C" void kernel_launch(void* const* d_in, const int* in_sizes, int n_in,
                              void* d_out, int out_size, void* d_ws, size_t ws_size,
                              hipStream_t stream) {
  const float* x     = (const float*)d_in[0];
  const float* W_DQ  = (const float*)d_in[1];
  const float* W_UQ  = (const float*)d_in[2];
  const float* W_QR  = (const float*)d_in[3];
  const float* W_DKV = (const float*)d_in[4];
  const float* W_UK  = (const float*)d_in[5];
  const float* W_UV  = (const float*)d_in[6];
  const float* W_KR  = (const float*)d_in[7];
  const float* W_O   = (const float*)d_in[8];
  float* out = (float*)d_out;

  char* w = (char*)d_ws;
  const size_t MiB = 1u << 20;
  unsigned short* xbf   = (unsigned short*)(w + 0   * MiB);  // 16 MiB [8192,1024]
  unsigned short* wts   = (unsigned short*)(w + 16  * MiB);  //  8 MiB region, transposed weights
  unsigned short* tDQ  = wts;                                //  [256,1024]   512 KiB
  unsigned short* tUQ  = wts + (256u * 1024u);               //  [1024,256]   512 KiB
  unsigned short* tQR  = tUQ + (1024u * 256u);               //  [512,256]    256 KiB
  unsigned short* tDKV = tQR + (512u * 256u);                //  [256,1024]   512 KiB
  unsigned short* tUK  = tDKV + (256u * 1024u);              //  [1024,256]   512 KiB
  unsigned short* tUV  = tUK + (1024u * 256u);               //  [1024,256]   512 KiB
  unsigned short* tKR  = tUV + (1024u * 256u);               //  [32,1024]     64 KiB
  unsigned short* tO   = tKR + (32u * 1024u);                //  [1024,1024]    2 MiB
  unsigned short* c_q  = (unsigned short*)(w + 24  * MiB);   //  4 MiB bf16 [8192,256]
  unsigned short* c_kv = (unsigned short*)(w + 28  * MiB);   //  4 MiB bf16 [8192,256]
  float* Qc  = (float*)(w + 32  * MiB);                      // 32 MiB [8192,1024]
  float* Qr  = (float*)(w + 64  * MiB);                      // 16 MiB [8192,512]
  float* Kc  = (float*)(w + 80  * MiB);                      // 32 MiB [8192,1024]
  float* Krp = (float*)(w + 112 * MiB);                      //  1 MiB [8192,32]
  float* Vf  = (float*)(w + 113 * MiB);                      // 32 MiB [8192,1024]
  unsigned short* Qb = (unsigned short*)(w + 145 * MiB);     // 24 MiB
  unsigned short* Kb = (unsigned short*)(w + 169 * MiB);     // 24 MiB
  unsigned short* Vt = (unsigned short*)(w + 193 * MiB);     // 16 MiB
  unsigned short* attnb = (unsigned short*)Qc;  // Qc dead after pack_q; 16 MiB bf16

  auto cast = [&](const float* s, unsigned short* d, size_t n) {
    cast_bf16_kernel<<<(int)((n + 255) / 256), 256, 0, stream>>>(s, d, n);
  };
  auto castT = [&](const float* s, unsigned short* d, int K, int N) {
    size_t n = (size_t)K * N;
    castT_bf16_kernel<<<(int)((n + 255) / 256), 256, 0, stream>>>(s, d, K, N);
  };
  auto gemm = [&](const unsigned short* A, const unsigned short* BT, void* C,
                  int M, int N, int K, int obf) {
    dim3 grid((N + GEMM_BN - 1) / GEMM_BN, M / GEMM_BM);
    gemm_bf16_kernel<<<grid, dim3(128), 0, stream>>>(A, BT, C, M, N, K, obf);
  };

  // One-time bf16 conversions (x row-major; weights transposed [N][K])
  cast(x, xbf, (size_t)MROWS * DMODEL);
  castT(W_DQ,  tDQ,  DMODEL, DLAT);
  castT(W_UQ,  tUQ,  DLAT,   NHEADS * DH);
  castT(W_QR,  tQR,  DLAT,   NHEADS * DHR);
  castT(W_DKV, tDKV, DMODEL, DLAT);
  castT(W_UK,  tUK,  DLAT,   NHEADS * DH);
  castT(W_UV,  tUV,  DLAT,   NHEADS * DH);
  castT(W_KR,  tKR,  DMODEL, DHR);
  castT(W_O,   tO,   NHEADS * DH, DMODEL);

  // Projections
  gemm(xbf,  tDQ,  c_q,  MROWS, DLAT,         DMODEL, 1);
  gemm(xbf,  tDKV, c_kv, MROWS, DLAT,         DMODEL, 1);
  gemm(xbf,  tKR,  Krp,  MROWS, DHR,          DMODEL, 0);
  gemm(c_q,  tUQ,  Qc,   MROWS, NHEADS * DH,  DLAT,   0);
  gemm(c_q,  tQR,  Qr,   MROWS, NHEADS * DHR, DLAT,   0);
  gemm(c_kv, tUK,  Kc,   MROWS, NHEADS * DH,  DLAT,   0);
  gemm(c_kv, tUV,  Vf,   MROWS, NHEADS * DH,  DLAT,   0);

  // RoPE + concat + bf16 packing
  {
    size_t tqk = (size_t)BDIM * NHEADS * SEQ * DQK;
    int blocks = (int)((tqk + 255) / 256);
    pack_q_kernel<<<blocks, 256, 0, stream>>>(Qc, Qr, Qb);
    pack_k_kernel<<<blocks, 256, 0, stream>>>(Kc, Krp, Kb);
    size_t tv = (size_t)BDIM * NHEADS * DH * SEQ;
    pack_vt_kernel<<<(int)((tv + 255) / 256), 256, 0, stream>>>(Vf, Vt);
  }

  // Causal flash attention (B*H*S/16 = 8192 wave-tasks, 4 waves/block)
  mla_attention_kernel<<<(BDIM * NHEADS * (SEQ / 16)) / 4, 128, 0, stream>>>(
      Qb, Kb, Vt, attnb);

  // Output projection (fp32 result)
  gemm(attnb, tO, out, MROWS, DMODEL, NHEADS * DH, 0);
}
